// NonCondWaveNet_76321568850678
// MI455X (gfx1250) — matrix operand order repaired and checked by reading, compiled, not run on hardware
//
#include <hip/hip_runtime.h>
#include <hip/hip_bf16.h>

typedef __attribute__((ext_vector_type(16))) _Float16 v16h;
typedef __attribute__((ext_vector_type(8)))  _Float16 v8h;
typedef __attribute__((ext_vector_type(8)))  float    v8f;

#define TT 32768
#define NB 8

#if __has_builtin(__builtin_amdgcn_global_load_async_to_lds_b128) && \
    __has_builtin(__builtin_amdgcn_s_wait_asynccnt)
#define WN_ASYNC_LDS 1
// exact GCC-vector pointee type per the builtin's signature:
//   void (int __vector(4) AS1*, int __vector(4) AS3*, imm int, imm int)
typedef int wn_v4i __attribute__((vector_size(16)));
typedef __attribute__((address_space(1))) wn_v4i as1_v4i;
typedef __attribute__((address_space(3))) wn_v4i as3_v4i;
#endif

static __device__ __forceinline__ v16h cat8(v8h lo, v8h hi) {
    return __builtin_shufflevector(lo, hi, 0,1,2,3,4,5,6,7,8,9,10,11,12,13,14,15);
}
static __device__ __forceinline__ v8f wmma(v16h a, v16h b, v8f c) {
    return __builtin_amdgcn_wmma_f32_16x16x32_f16(false, a, false, b, (short)0, c, false, false);
}

#define WN_LOG2E 1.4426950408889634f

static __device__ __forceinline__ float fast_exp2(float x) {
#if __has_builtin(__builtin_amdgcn_exp2f)
    return __builtin_amdgcn_exp2f(x);
#else
    return exp2f(x);
#endif
}
static __device__ __forceinline__ float fast_rcp(float x) {
#if __has_builtin(__builtin_amdgcn_rcpf)
    return __builtin_amdgcn_rcpf(x);
#else
    return 1.0f / x;
#endif
}
// tanh(f)*sigmoid(g) via v_exp_f32 / v_rcp_f32 only (no divides, no branches)
static __device__ __forceinline__ float gated(float f, float g) {
    float e2f = fast_exp2(2.0f * WN_LOG2E * f);            // e^(2f)
    float th  = 1.0f - 2.0f * fast_rcp(e2f + 1.0f);        // tanh(f)
    float sg  = fast_rcp(1.0f + fast_exp2(-WN_LOG2E * g)); // sigmoid(g)
    return th * sg;
}

// ---------------- conversion kernels ----------------

__global__ __launch_bounds__(256) void wn_convert_x(const float* __restrict__ x,
                                                    _Float16* __restrict__ x16) {
    size_t i = (size_t)blockIdx.x * 256 + threadIdx.x;
    x16[i] = (_Float16)x[i];
}

__global__ __launch_bounds__(256) void wn_convert_w(
    const float* __restrict__ wdil, const float* __restrict__ wres,
    const float* __restrict__ wskip, const float* __restrict__ wpre,
    const float* __restrict__ wfin,
    _Float16* __restrict__ Wd, _Float16* __restrict__ Wr, _Float16* __restrict__ Wsk,
    _Float16* __restrict__ Wp, _Float16* __restrict__ Wf) {
    int idx = blockIdx.x * 256 + threadIdx.x;
    if (idx < 491520) {                       // w_dil [30,2,64,128] -> [30][cout=128][k=tap*64+cin]
        int i = idx / 16384, rem = idx % 16384;
        int cout = rem & 127, k = rem >> 7;
        int tap = k >> 6, cin = k & 63;
        Wd[(size_t)i * 16384 + cout * 128 + k] =
            (_Float16)wdil[(((size_t)i * 2 + tap) * 64 + cin) * 128 + cout];
        return;
    }
    idx -= 491520;
    if (idx < 122880) {                       // w_res [30,1,64,64] -> [30][cout][cin]
        int i = idx / 4096, rem = idx % 4096;
        int cout = rem & 63, cin = rem >> 6;
        Wr[(size_t)i * 4096 + cout * 64 + cin] = (_Float16)wres[((size_t)i * 64 + cin) * 64 + cout];
        return;
    }
    idx -= 122880;
    if (idx < 122880) {                       // w_skip
        int i = idx / 4096, rem = idx % 4096;
        int cout = rem & 63, cin = rem >> 6;
        Wsk[(size_t)i * 4096 + cout * 64 + cin] = (_Float16)wskip[((size_t)i * 64 + cin) * 64 + cout];
        return;
    }
    idx -= 122880;
    if (idx < 4096) {                         // w_pre [1,64,64] -> [cout][cin]
        int cout = idx & 63, cin = idx >> 6;
        Wp[cout * 64 + cin] = (_Float16)wpre[cin * 64 + cout];
        return;
    }
    idx -= 4096;
    if (idx < 16384) {                        // w_fin [1,64,256] -> [cout=256][cin]
        int cout = idx & 255, cin = idx >> 8;
        Wf[cout * 64 + cin] = (_Float16)wfin[cin * 256 + cout];
    }
}

// ---------------- per-layer fused kernel ----------------
// block: 128 timesteps, 256 threads (8 waves)

__global__ __launch_bounds__(256) void wn_layer(
    const _Float16* __restrict__ xin, _Float16* __restrict__ xout,
    float* __restrict__ skip,
    const _Float16* __restrict__ Wd, const float* __restrict__ bdil,
    const _Float16* __restrict__ Wr, const float* __restrict__ bres,
    const _Float16* __restrict__ Wsk, const float* __restrict__ bskip,
    int dil, int first) {
    __shared__ __align__(16) _Float16 At[128 * 136]; // [m][ x[t-dil](64) | x[t](64) ]
    __shared__ __align__(16) _Float16 Zt[128 * 72];  // gated activation

    const int b  = blockIdx.y;
    const int t0 = blockIdx.x * 128;
    const int tid = threadIdx.x;

    // ---- load A-tile (async LDS DMA when available) ----
    {
        int m = tid >> 1, half = tid & 1;
        int ts = t0 + m - (half ? 0 : dil);
        _Float16* dst = &At[m * 136 + half * 64];
        if (ts >= 0) {
            const _Float16* src = xin + ((size_t)b * TT + ts) * 64;
#ifdef WN_ASYNC_LDS
            #pragma unroll
            for (int j = 0; j < 8; ++j)
                __builtin_amdgcn_global_load_async_to_lds_b128(
                    (as1_v4i*)(src + j * 8), (as3_v4i*)(dst + j * 8), 0, 0);
#else
            #pragma unroll
            for (int j = 0; j < 8; ++j) ((uint4*)dst)[j] = ((const uint4*)src)[j];
#endif
        } else {
            uint4 z = {0, 0, 0, 0};
            #pragma unroll
            for (int j = 0; j < 8; ++j) ((uint4*)dst)[j] = z;
        }
    }
#ifdef WN_ASYNC_LDS
    __builtin_amdgcn_s_wait_asynccnt(0);
#endif
    __syncthreads();

    const int wave = tid >> 5, lane = tid & 31;
    const int n16 = lane & 15, hh = lane >> 4;

    // ---- dilated conv GEMM [128x128]x[128->128], f/g tile pairs per wave ----
    for (int job = wave; job < 32; job += 8) {
        int rt = job >> 2, cf = job & 3, cg = cf + 4;
        v8f accF = {}, accG = {};
        const int mrow = rt * 16 + n16;
        #pragma unroll
        for (int ks = 0; ks < 4; ++ks) {
            const _Float16* ap = &At[mrow * 136 + ks * 32 + hh * 16];
            v16h a = cat8(*(const v8h*)ap, *(const v8h*)(ap + 8));
            v16h bf = *(const v16h*)(Wd + (size_t)(cf * 16 + n16) * 128 + ks * 32 + hh * 16);
            v16h bg = *(const v16h*)(Wd + (size_t)(cg * 16 + n16) * 128 + ks * 32 + hh * 16);
            accF = wmma(a, bf, accF);
            accG = wmma(a, bg, accG);
        }
        float bf_ = bdil[cf * 16 + n16];
        float bg_ = bdil[64 + cf * 16 + n16];
        #pragma unroll
        for (int r = 0; r < 8; ++r) {
            float z = gated(accF[r] + bf_, accG[r] + bg_);
            Zt[(rt * 16 + hh * 8 + r) * 72 + cf * 16 + n16] = (_Float16)z;
        }
    }
    __syncthreads();

    // ---- skip & residual GEMMs [128x64]x[64->64] ----
    for (int job = wave; job < 64; job += 8) {
        int mat = job & 1, ct = (job >> 1) & 3, rt = job >> 3;
        const _Float16* W = mat ? Wsk : Wr;
        v8f acc = {};
        const int mrow = rt * 16 + n16;
        #pragma unroll
        for (int ks = 0; ks < 2; ++ks) {
            const _Float16* ap = &Zt[mrow * 72 + ks * 32 + hh * 16];
            v16h a = cat8(*(const v8h*)ap, *(const v8h*)(ap + 8));
            v16h bm = *(const v16h*)(W + (size_t)(ct * 16 + n16) * 64 + ks * 32 + hh * 16);
            acc = wmma(a, bm, acc);
        }
        int ch = ct * 16 + n16;
        if (mat == 0) {
            float bb = bres[ch];
            #pragma unroll
            for (int r = 0; r < 8; ++r) {
                int m = rt * 16 + hh * 8 + r;
                float val = acc[r] + bb + (float)At[m * 136 + 64 + ch];
                xout[((size_t)b * TT + t0 + m) * 64 + ch] = (_Float16)val;
            }
        } else {
            float bb = bskip[ch];
            #pragma unroll
            for (int r = 0; r < 8; ++r) {
                int m = rt * 16 + hh * 8 + r;
                size_t gi = ((size_t)b * TT + t0 + m) * 64 + ch;
                float val = acc[r] + bb + (first ? 0.0f : skip[gi]);
                skip[gi] = val;
            }
        }
    }
}

// ---------------- head: relu -> 1x1 -> relu -> 1x1 -> softmax ----------------
// block: 32 timesteps, 256 threads (8 waves)

__global__ __launch_bounds__(256) void wn_final(
    const float* __restrict__ skip,
    const _Float16* __restrict__ Wp, const float* __restrict__ bpre,
    const _Float16* __restrict__ Wf, const float* __restrict__ bfin,
    float* __restrict__ out) {
    __shared__ __align__(16) _Float16 A1[32 * 72];
    __shared__ __align__(16) _Float16 A2[32 * 72];
    __shared__ float    Y[32 * 256];
    __shared__ float    red[256];

    const int b  = blockIdx.y;
    const int t0 = blockIdx.x * 32;
    const int tid = threadIdx.x;

    for (int e = tid; e < 2048; e += 256) {
        int m = e >> 6, c = e & 63;
        float v = skip[((size_t)b * TT + t0 + m) * 64 + c];
        A1[m * 72 + c] = (_Float16)fmaxf(v, 0.0f);
    }
    __syncthreads();

    const int wave = tid >> 5, lane = tid & 31;
    const int n16 = lane & 15, hh = lane >> 4;

    // GEMM1: [32x64]x[64->64], relu
    for (int job = wave; job < 8; job += 8) {
        int rt = job >> 2, ct = job & 3;
        v8f acc = {};
        const int mrow = rt * 16 + n16;
        #pragma unroll
        for (int ks = 0; ks < 2; ++ks) {
            const _Float16* ap = &A1[mrow * 72 + ks * 32 + hh * 16];
            v16h a = cat8(*(const v8h*)ap, *(const v8h*)(ap + 8));
            v16h bm = *(const v16h*)(Wp + (size_t)(ct * 16 + n16) * 64 + ks * 32 + hh * 16);
            acc = wmma(a, bm, acc);
        }
        float bb = bpre[ct * 16 + n16];
        #pragma unroll
        for (int r = 0; r < 8; ++r)
            A2[(rt * 16 + hh * 8 + r) * 72 + ct * 16 + n16] = (_Float16)fmaxf(acc[r] + bb, 0.0f);
    }
    __syncthreads();

    // GEMM2: [32x64]x[64->256]
    for (int job = wave; job < 32; job += 8) {
        int rt = job >> 4, ct = job & 15;
        v8f acc = {};
        const int mrow = rt * 16 + n16;
        #pragma unroll
        for (int ks = 0; ks < 2; ++ks) {
            const _Float16* ap = &A2[mrow * 72 + ks * 32 + hh * 16];
            v16h a = cat8(*(const v8h*)ap, *(const v8h*)(ap + 8));
            v16h bm = *(const v16h*)(Wf + (size_t)(ct * 16 + n16) * 64 + ks * 32 + hh * 16);
            acc = wmma(a, bm, acc);
        }
        float bb = bfin[ct * 16 + n16];
        #pragma unroll
        for (int r = 0; r < 8; ++r)
            Y[(rt * 16 + hh * 8 + r) * 256 + ct * 16 + n16] = acc[r] + bb;
    }
    __syncthreads();

    // softmax: 8 threads per row, 32 elements each
    const int row = tid >> 3, part = tid & 7;
    float* yrow = &Y[row * 256 + part * 32];
    float lmax = -1e30f;
    #pragma unroll 8
    for (int j = 0; j < 32; ++j) lmax = fmaxf(lmax, yrow[j]);
    red[tid] = lmax;
    __syncthreads();
    float rmax = -1e30f;
    #pragma unroll
    for (int p = 0; p < 8; ++p) rmax = fmaxf(rmax, red[row * 8 + p]);
    __syncthreads();
    float lsum = 0.0f;
    #pragma unroll 8
    for (int j = 0; j < 32; ++j) {
        float e = fast_exp2(WN_LOG2E * (yrow[j] - rmax));
        yrow[j] = e;
        lsum += e;
    }
    red[tid] = lsum;
    __syncthreads();
    float rsum = 0.0f;
    #pragma unroll
    for (int p = 0; p < 8; ++p) rsum += red[row * 8 + p];
    float inv = fast_rcp(rsum);
    float* optr = out + ((size_t)b * TT + t0 + row) * 256 + part * 32;
    #pragma unroll 8
    for (int j = 0; j < 32; ++j) optr[j] = yrow[j] * inv;
}

// ---------------- launcher ----------------

extern "C" void kernel_launch(void* const* d_in, const int* in_sizes, int n_in,
                              void* d_out, int out_size, void* d_ws, size_t ws_size,
                              hipStream_t stream) {
    const float* x     = (const float*)d_in[0];
    const float* wdil  = (const float*)d_in[1];
    const float* bdil  = (const float*)d_in[2];
    const float* wres  = (const float*)d_in[3];
    const float* bres  = (const float*)d_in[4];
    const float* wskip = (const float*)d_in[5];
    const float* bskip = (const float*)d_in[6];
    const float* wpre  = (const float*)d_in[7];
    const float* bpre  = (const float*)d_in[8];
    const float* wfin  = (const float*)d_in[9];
    const float* bfin  = (const float*)d_in[10];

    char* ws = (char*)d_ws;
    size_t off = 0;
    auto take = [&](size_t bytes) { size_t o = off; off += (bytes + 255) & ~(size_t)255; return o; };
    _Float16* Wd  = (_Float16*)(ws + take(491520ull * 2));
    _Float16* Wr  = (_Float16*)(ws + take(122880ull * 2));
    _Float16* Wsk = (_Float16*)(ws + take(122880ull * 2));
    _Float16* Wp  = (_Float16*)(ws + take(4096ull * 2));
    _Float16* Wf  = (_Float16*)(ws + take(16384ull * 2));
    _Float16* xA  = (_Float16*)(ws + take((size_t)NB * TT * 64 * 2));
    _Float16* xB  = (_Float16*)(ws + take((size_t)NB * TT * 64 * 2));
    float*    skp = (float*)   (ws + take((size_t)NB * TT * 64 * 4));

    wn_convert_x<<<(NB * TT * 64) / 256, 256, 0, stream>>>(x, xA);
    wn_convert_w<<<2960, 256, 0, stream>>>(wdil, wres, wskip, wpre, wfin, Wd, Wr, Wsk, Wp, Wf);

    const _Float16* cur = xA;
    _Float16* nxt = xB;
    for (int i = 0; i < 30; ++i) {
        int dil = 1 << (i % 10);
        wn_layer<<<dim3(TT / 128, NB), 256, 0, stream>>>(
            cur, nxt, skp,
            Wd + (size_t)i * 16384, bdil + (size_t)i * 128,
            Wr + (size_t)i * 4096,  bres + (size_t)i * 64,
            Wsk + (size_t)i * 4096, bskip + (size_t)i * 64,
            dil, i == 0 ? 1 : 0);
        const _Float16* t = cur; cur = nxt; nxt = (_Float16*)t;
    }

    wn_final<<<dim3(TT / 32, NB), 256, 0, stream>>>(skp, Wp, bpre, Wf, bfin, (float*)d_out);
}
